// BitConv_72353019069105
// MI455X (gfx1250) — matrix operand order repaired and checked by reading, not compiled, not run on hardware
//
#include <hip/hip_runtime.h>
#include <math.h>
#include <stdint.h>

typedef __attribute__((ext_vector_type(16))) _Float16 v16h;
typedef __attribute__((ext_vector_type(8)))  _Float16 v8h;
typedef __attribute__((ext_vector_type(8)))  float    v8f;
typedef __attribute__((ext_vector_type(4)))  int      v4i;

#define C1   256
#define C2   256
#define HH   56
#define WW   56
#define BB   32
#define KTOT 2304                 // C1 * 9
#define NTOT (BB * HH * WW)       // 100352 = 784 * 128
#define NWEI (C2 * KTOT)          // 589824

// ---- workspace layout (bytes) ----
#define WS_HIST   0      // 256 x u32
#define WS_PREFIX 1024   // u32
#define WS_KREM   1028   // u32
#define WS_R0     1032   // f32 (k-th value, rank n/2-1)
#define WS_R1     1036   // f32 (k-th value, rank n/2)
#define WS_SCALE  1088   // 256 x f32 : m * gamma / sqrt(var+eps)
#define WS_SHIFT  2112   // 256 x f32 : beta - mean * inv
#define WS_WQ     3200   // 589824 x f16 ternary weights, [C2][C1*9] row-major

// ================= median via radix select =================

__global__ void init_select(unsigned* ws_u, unsigned ktarget) {
    if (threadIdx.x < 256) ws_u[WS_HIST / 4 + threadIdx.x] = 0u;
    if (threadIdx.x == 0) {
        ws_u[WS_PREFIX / 4] = 0u;
        ws_u[WS_KREM / 4]   = ktarget;
    }
}

__global__ void hist_pass(const float* __restrict__ w, int n,
                          unsigned* __restrict__ ws_u, int pass) {
    const unsigned prefix = ws_u[WS_PREFIX / 4];
    const int shift = 24 - 8 * pass;
    const unsigned hmask = (pass == 0) ? 0u : (0xFFFFFFFFu << (shift + 8));
    for (int i = blockIdx.x * blockDim.x + threadIdx.x; i < n;
         i += gridDim.x * blockDim.x) {
        unsigned bits = __float_as_uint(fabsf(w[i]));
        if (((bits ^ prefix) & hmask) == 0u)
            atomicAdd(&ws_u[WS_HIST / 4 + ((bits >> shift) & 0xFFu)], 1u);
    }
}

__global__ void select_pass(unsigned* ws_u, float* ws_f, int pass, int res_slot) {
    if (threadIdx.x != 0) return;
    const int shift = 24 - 8 * pass;
    unsigned krem   = ws_u[WS_KREM / 4];
    unsigned prefix = ws_u[WS_PREFIX / 4];
    unsigned cum = 0, sel = 0;
    bool found = false;
    for (int b = 0; b < 256; ++b) {
        unsigned c = ws_u[WS_HIST / 4 + b];
        ws_u[WS_HIST / 4 + b] = 0u;           // re-zero for next pass
        if (!found) {
            if (krem < cum + c) { found = true; sel = (unsigned)b; krem -= cum; }
            else cum += c;
        }
    }
    prefix |= sel << shift;
    ws_u[WS_PREFIX / 4] = prefix;
    ws_u[WS_KREM / 4]   = krem;
    if (pass == 3) ws_f[res_slot] = __uint_as_float(prefix);
}

// ================= BN folding + weight ternarization =================

__global__ void finalize_bn(const float* __restrict__ gamma,
                            const float* __restrict__ beta,
                            const float* __restrict__ rmean,
                            const float* __restrict__ rvar,
                            float* __restrict__ ws_f) {
    int c = blockIdx.x * blockDim.x + threadIdx.x;
    if (c >= C2) return;
    float med  = 0.5f * (ws_f[WS_R0 / 4] + ws_f[WS_R1 / 4]);
    float invs = fmaxf(med, 1e-5f);                       // 1/scale
    float inv  = gamma[c] * rsqrtf(rvar[c] + 1e-5f);
    ws_f[WS_SCALE / 4 + c] = invs * inv;                  // folds median into BN
    ws_f[WS_SHIFT / 4 + c] = beta[c] - rmean[c] * inv;
}

__global__ void ternarize_w(const float* __restrict__ w,
                            const float* __restrict__ ws_f,
                            _Float16* __restrict__ wq) {
    int i = blockIdx.x * blockDim.x + threadIdx.x;
    if (i >= NWEI) return;
    float invs  = fmaxf(0.5f * (ws_f[WS_R0 / 4] + ws_f[WS_R1 / 4]), 1e-5f);
    float scale = 1.0f / invs;
    float t = fminf(fmaxf(rintf(w[i] * scale), -1.0f), 1.0f);  // round-half-even
    wq[i] = (_Float16)t;   // exact in f16: -1, 0, +1
}

// ================= implicit-GEMM conv with WMMA =================

#define BM  128
#define BN  128
#define BKc 64   // two WMMA k-steps per LDS stage
#define LDA 72   // f16 elems; 144 B row stride -> all frag reads 16B-aligned
#define LDB 72

__device__ __forceinline__ v16h cat8(v8h lo, v8h hi) {
    return __builtin_shufflevector(lo, hi, 0, 1, 2, 3, 4, 5, 6, 7,
                                           8, 9, 10, 11, 12, 13, 14, 15);
}

#if __has_builtin(__builtin_amdgcn_global_load_async_to_lds_b128)
#define USE_ASYNC_LDS 1
typedef __attribute__((address_space(1))) v4i g_v4i;   // global int4
typedef __attribute__((address_space(3))) v4i l_v4i;   // LDS int4
#endif

__device__ __forceinline__ void wait_async_then_barrier() {
#if defined(USE_ASYNC_LDS)
#if __has_builtin(__builtin_amdgcn_s_wait_asynccnt)
    __builtin_amdgcn_s_wait_asynccnt(0);
#else
    asm volatile("s_wait_asynccnt 0x0" ::: "memory");
#endif
#endif
    __syncthreads();
}

__global__ __launch_bounds__(256)
void bitconv_wmma(const float* __restrict__ x, const _Float16* __restrict__ wq,
                  const float* __restrict__ scale_t,
                  const float* __restrict__ shift_t,
                  float* __restrict__ out) {
    __shared__ __align__(16) _Float16 sA[BM * LDA];
    __shared__ __align__(16) _Float16 sB[BN * LDB];

    const int tid  = threadIdx.x;
    const int lane = tid & 31;
    const int wave = tid >> 5;
    const int wm   = wave >> 2;        // 0..1  -> 64 rows of M
    const int wn   = wave & 3;         // 0..3  -> 32 cols of N
    const int m0   = blockIdx.y * BM;
    const int n0   = blockIdx.x * BN;

    // B-gather mapping: this thread owns pixel p, k-rows kkbase + 2*it
    const int p      = tid & 127;
    const int kkbase = tid >> 7;       // 0 or 1
    const int n_g    = n0 + p;
    const int b_idx  = n_g / (HH * WW);
    const int hw     = n_g - b_idx * (HH * WW);
    const int h      = hw / WW;
    const int wpix   = hw - h * WW;
    const float* xb  = x + (size_t)b_idx * (C1 * HH * WW);

    v8f acc[4][2] = {};

    for (int kt = 0; kt < KTOT; kt += BKc) {
        // ---- A tile: 128 x 64 f16; async DMA to LDS (ASYNCcnt) or sync copy ----
        #pragma unroll
        for (int it = 0; it < 4; ++it) {
            int e   = tid + it * 256;
            int row = e >> 3;
            int c8  = e & 7;
            const _Float16* gsrc = wq + (size_t)(m0 + row) * KTOT + kt + c8 * 8;
            _Float16*       ldst = &sA[row * LDA + c8 * 8];
#if defined(USE_ASYNC_LDS)
            __builtin_amdgcn_global_load_async_to_lds_b128(
                (g_v4i*)(const void*)gsrc, (l_v4i*)(void*)ldst, 0, 0);
#else
            *(uint4*)ldst = *(const uint4*)gsrc;
#endif
        }
        if (kt + BKc < KTOT)   // gfx1250 global_prefetch_b8 for next weight tile
            __builtin_prefetch(wq + (size_t)m0 * KTOT + kt + BKc, 0, 0);

        // ---- B tile: im2col gather, stored transposed sB[p][kk] ----
        #pragma unroll
        for (int it = 0; it < 32; ++it) {
            int kk = kkbase + 2 * it;
            int k  = kt + kk;
            int c  = k / 9;
            int rs = k - c * 9;
            int r  = rs / 3;
            int s  = rs - r * 3;
            int ih = h + r - 1;
            int iw = wpix + s - 1;
            float v = 0.0f;
            if ((unsigned)ih < (unsigned)HH && (unsigned)iw < (unsigned)WW)
                v = xb[(size_t)c * (HH * WW) + ih * WW + iw];
            sB[p * LDB + kk] = (_Float16)v;
        }
        wait_async_then_barrier();

        // ---- fragments per ISA 16-bit A/B layouts, 2 k-steps x (4x2) WMMA ----
        const int mrow = lane & 15;
        const int half = lane >> 4;
        #pragma unroll
        for (int ks = 0; ks < 2; ++ks) {
            v16h af[4], bf[2];
            #pragma unroll
            for (int i = 0; i < 4; ++i) {
                const _Float16* pa =
                    &sA[(wm * 64 + i * 16 + mrow) * LDA + ks * 32 + half * 8];
                af[i] = cat8(*(const v8h*)(pa), *(const v8h*)(pa + 16));
            }
            #pragma unroll
            for (int j = 0; j < 2; ++j) {
                const _Float16* pb =
                    &sB[(wn * 32 + j * 16 + mrow) * LDB + ks * 32 + half * 8];
                bf[j] = cat8(*(const v8h*)(pb), *(const v8h*)(pb + 16));
            }
            #pragma unroll
            for (int i = 0; i < 4; ++i)
                #pragma unroll
                for (int j = 0; j < 2; ++j)
                    acc[i][j] = __builtin_amdgcn_wmma_f32_16x16x32_f16(
                        false, af[i], false, bf[j], (short)0, acc[i][j],
                        false, false);
        }
        __syncthreads();
    }

    // ---- epilogue: folded BN (incl. median) + SiLU, NCHW store ----
    const int mrow = lane & 15;
    const int half = lane >> 4;
    #pragma unroll
    for (int j = 0; j < 2; ++j) {
        int n_out   = n0 + wn * 32 + j * 16 + mrow;
        int bo      = n_out / (HH * WW);
        int hwo     = n_out - bo * (HH * WW);
        size_t base = (size_t)bo * C2 * (HH * WW) + hwo;
        #pragma unroll
        for (int i = 0; i < 4; ++i) {
            #pragma unroll
            for (int v = 0; v < 8; ++v) {
                int c2 = m0 + wm * 64 + i * 16 + half * 8 + v;
                float val = acc[i][j][v] * scale_t[c2] + shift_t[c2];
                float sig = 1.0f / (1.0f + __expf(-val));
                out[base + (size_t)c2 * (HH * WW)] = val * sig;
            }
        }
    }
}

// ================= host launcher =================

extern "C" void kernel_launch(void* const* d_in, const int* in_sizes, int n_in,
                              void* d_out, int out_size, void* d_ws, size_t ws_size,
                              hipStream_t stream) {
    (void)in_sizes; (void)n_in; (void)out_size; (void)ws_size;
    const float* x     = (const float*)d_in[0];
    const float* w     = (const float*)d_in[1];
    const float* gamma = (const float*)d_in[2];
    const float* beta  = (const float*)d_in[3];
    const float* rmean = (const float*)d_in[4];
    const float* rvar  = (const float*)d_in[5];
    float* out = (float*)d_out;

    char*     ws   = (char*)d_ws;
    unsigned* ws_u = (unsigned*)ws;
    float*    ws_f = (float*)ws;
    _Float16* wq   = (_Float16*)(ws + WS_WQ);

    // exact median of |w| (even count -> mean of the two middle order stats)
    for (int t = 0; t < 2; ++t) {
        unsigned ktarget = (unsigned)(NWEI / 2 - 1 + t);
        init_select<<<1, 256, 0, stream>>>(ws_u, ktarget);
        for (int pass = 0; pass < 4; ++pass) {
            hist_pass<<<512, 256, 0, stream>>>(w, NWEI, ws_u, pass);
            select_pass<<<1, 64, 0, stream>>>(ws_u, ws_f, pass, WS_R0 / 4 + t);
        }
    }
    finalize_bn<<<1, 256, 0, stream>>>(gamma, beta, rmean, rvar, ws_f);
    ternarize_w<<<(NWEI + 255) / 256, 256, 0, stream>>>(w, ws_f, wq);

    dim3 grid(NTOT / BN, C2 / BM);   // 784 x 2
    bitconv_wmma<<<grid, 256, 0, stream>>>(x, wq,
                                           ws_f + WS_SCALE / 4,
                                           ws_f + WS_SHIFT / 4, out);
}
